// Decoder_50294067036734
// MI455X (gfx1250) — compile-verified, matrix operand order checked by reading, not run on hardware
//
#include <hip/hip_runtime.h>
#include <hip/hip_bf16.h>
#include <math.h>

// ---------------- problem constants (match reference) ----------------
#define V_TOT  50000
#define VG_    20000
#define H_     512
#define DW_    256
#define BEAM_  4
#define STEPS_ 16
#define BOS_   2
#define TP_    400
#define B_     8
#define BB_    32            // B * BEAM

typedef __attribute__((ext_vector_type(16))) _Float16 v16h;
typedef __attribute__((ext_vector_type(8)))  _Float16 v8h;
typedef __attribute__((ext_vector_type(8)))  float    v8f;

// ============================================================================
// Weight packing into WMMA B-operand layout.
// For each (ktile of 32, ntile of 16): lane L owns 16 contiguous halfs.
//   lane<16 : n = nt*16+L,      halfs j<8 -> K=kt*32+j,    j>=8 -> K=kt*32+16..23
//   lane>=16: n = nt*16+L-16,   halfs j<8 -> K=kt*32+8..15, j>=8 -> K=kt*32+24..31
// srcT==0: src is [K,N] row-major.  srcT==1: src is [N,K] row-major (i.e. W.T).
// ============================================================================
__global__ void pack_b_kernel(const float* __restrict__ src, _Float16* __restrict__ dst,
                              int K, int N, int srcT)
{
    const int nt = blockIdx.x, kt = blockIdx.y, lane = threadIdx.x;
    const int half = lane >> 4, lm = lane & 15;
    const int n = nt * 16 + lm;
    _Float16* out = dst + ((((size_t)kt * gridDim.x + nt) << 5) + lane) * 16;
#pragma unroll
    for (int j = 0; j < 16; ++j) {
        const int k = kt * 32 + half * 8 + (j < 8 ? j : j + 8);
        const float v = srcT ? src[(size_t)n * K + k] : src[(size_t)k * N + n];
        out[j] = (_Float16)v;
    }
}

__global__ void f32_to_f16_kernel(const float* __restrict__ src, _Float16* __restrict__ dst, int n)
{
    const int i = blockIdx.x * 256 + threadIdx.x;
    if (i < n) dst[i] = (_Float16)src[i];
}

// ============================================================================
// Skinny GEMM: one wave per 32x16 C strip (two 16x16 tiles share each B load).
// C[M,N] (f32) = A[M,K] (f16 row-major, lda) x Bpacked[K,N]  [+bias] [+=C] [tanh] [->f16]
// flags: 1=accumulate into C, 2=tanh epilogue, 4=add bias[n], 8=also store f16 to Ch
// grid = (N/16, M/32), block = 32 (one wave).  M multiple of 32, N of 16, K of 32.
// ============================================================================
__global__ void wmma_gemm_kernel(const _Float16* __restrict__ A, int lda,
                                 const _Float16* __restrict__ Bp,
                                 const float* __restrict__ bias,
                                 float* __restrict__ C, int ldc,
                                 int K, int flags, _Float16* __restrict__ Ch)
{
    const int lane = threadIdx.x;
    const int nt = blockIdx.x, ms = blockIdx.y, ntiles = gridDim.x;
    const int half = lane >> 4, lm = lane & 15;
    const _Float16* arow0 = A + (size_t)(ms * 32 + lm) * lda;       // rows  ms*32 .. +15
    const _Float16* arow1 = arow0 + (size_t)16 * lda;               // rows  ms*32+16 .. +31

    v8f acc0 = {}, acc1 = {};
    const int nkt = K >> 5;
    for (int kt = 0; kt < nkt; ++kt) {
        const int k0 = (kt << 5) + (half << 3);
        // A operands: 16-bit A 16x32 layout (two 128b contiguous loads per tile)
        v8h a00 = *(const v8h*)(arow0 + k0);
        v8h a01 = *(const v8h*)(arow0 + k0 + 16);
        v8h a10 = *(const v8h*)(arow1 + k0);
        v8h a11 = *(const v8h*)(arow1 + k0 + 16);
        v16h aA, aB;
#pragma unroll
        for (int i = 0; i < 8; ++i) {
            aA[i] = a00[i]; aA[i + 8] = a01[i];
            aB[i] = a10[i]; aB[i + 8] = a11[i];
        }
        // B operand: pre-packed, 16 contiguous halfs per lane; shared by both tiles
        const _Float16* bptr = Bp + ((((size_t)kt * ntiles + nt) << 5) + lane) * 16;
        __builtin_prefetch(bptr + ((size_t)ntiles << 5) * 16, 0, 3);
        v16h b = *(const v16h*)bptr;
        acc0 = __builtin_amdgcn_wmma_f32_16x16x32_f16(false, aA, false, b,
                                                      (short)0, acc0, false, false);
        acc1 = __builtin_amdgcn_wmma_f32_16x16x32_f16(false, aB, false, b,
                                                      (short)0, acc1, false, false);
    }

    // C/D layout: VGPR r -> M = tile_base + half*8 + r ; N = nt*16 + lm
    const int n = nt * 16 + lm;
    const float bv = (flags & 4) ? bias[n] : 0.f;
#pragma unroll
    for (int r = 0; r < 8; ++r) {
        const int mm0 = ms * 32 + half * 8 + r;
        const int mm1 = mm0 + 16;
        const size_t i0 = (size_t)mm0 * ldc + n;
        const size_t i1 = (size_t)mm1 * ldc + n;
        float v0 = acc0[r] + bv;
        float v1 = acc1[r] + bv;
        if (flags & 1) { v0 += C[i0]; v1 += C[i1]; }
        if (flags & 2) { v0 = tanhf(v0); v1 = tanhf(v1); }
        C[i0] = v0; C[i1] = v1;
        if (flags & 8) { Ch[i0] = (_Float16)v0; Ch[i1] = (_Float16)v1; }
    }
}

// ============================================================================
// State init: h0 = tile(enc_hidden), c0 = 0, cov0 = 0, tok0 = BOS
// ============================================================================
__global__ void init_state_kernel(const float* __restrict__ enc_hidden,
                                  float* h, _Float16* h_h, float* c,
                                  float* cov, int* tok)
{
    const int bb = blockIdx.x, i = threadIdx.x;         // block 512
    const float hv = enc_hidden[(size_t)(bb & 7) * H_ + i];
    h[bb * H_ + i] = hv;
    h_h[bb * H_ + i] = (_Float16)hv;
    c[bb * H_ + i] = 0.f;
    if (i < TP_) cov[bb * TP_ + i] = 0.f;
    if (i == 0) tok[bb] = BOS_;
}

__global__ void embed_kernel(const int* __restrict__ tok, const float* __restrict__ E,
                             float* __restrict__ emb, _Float16* __restrict__ emb_h)
{
    const int bb = blockIdx.x, d = threadIdx.x;         // block 256
    const float v = E[(size_t)tok[bb] * DW_ + d];
    emb[bb * DW_ + d] = v;
    emb_h[bb * DW_ + d] = (_Float16)v;
}

// LSTM cell, torch gate order i,f,g,o
__global__ void lstm_cell_kernel(const float* __restrict__ gates,
                                 const float* __restrict__ b_ih, const float* __restrict__ b_hh,
                                 float* __restrict__ h, _Float16* __restrict__ h_h,
                                 float* __restrict__ c)
{
    const int bb = blockIdx.x, j = threadIdx.x;         // block 512
    const float* g = gates + (size_t)bb * 4 * H_;
    const float gi = g[j]            + b_ih[j]            + b_hh[j];
    const float gf = g[H_ + j]       + b_ih[H_ + j]       + b_hh[H_ + j];
    const float gg = g[2 * H_ + j]   + b_ih[2 * H_ + j]   + b_hh[2 * H_ + j];
    const float go = g[3 * H_ + j]   + b_ih[3 * H_ + j]   + b_hh[3 * H_ + j];
    const float si = 1.f / (1.f + expf(-gi));
    const float sf = 1.f / (1.f + expf(-gf));
    const float so = 1.f / (1.f + expf(-go));
    const float cn = sf * c[bb * H_ + j] + si * tanhf(gg);
    const float hn = so * tanhf(cn);
    c[bb * H_ + j] = cn;
    h[bb * H_ + j] = hn;
    h_h[bb * H_ + j] = (_Float16)hn;
}

// e[bb,t] = sum_h tanh(q + key + cov*w_cov) * v_att  -- grid (TP_, BB_), block 256
__global__ void attn_energy_kernel(const float* __restrict__ q, const float* __restrict__ keys,
                                   const float* __restrict__ cov, const float* __restrict__ w_cov,
                                   const float* __restrict__ v_att, float* __restrict__ e)
{
    const int t = blockIdx.x, bb = blockIdx.y, b = bb & 7, tid = threadIdx.x;
    const float* qr = q + (size_t)bb * H_;
    const float* kr = keys + ((size_t)(t * B_ + b)) * H_;
    const float cv = cov[bb * TP_ + t];
    float s = 0.f;
    for (int hh = tid; hh < H_; hh += 256)
        s += tanhf(qr[hh] + kr[hh] + cv * w_cov[hh]) * v_att[hh];
    __shared__ float sm[256];
    sm[tid] = s; __syncthreads();
    for (int o = 128; o > 0; o >>= 1) { if (tid < o) sm[tid] += sm[tid + o]; __syncthreads(); }
    if (tid == 0) e[bb * TP_ + t] = sm[0];
}

// softmax over Tp + coverage update; in-place e -> attn. grid 32, block 512
__global__ void attn_softmax_kernel(float* __restrict__ e, float* __restrict__ cov)
{
    const int bb = blockIdx.x, tid = threadIdx.x;
    __shared__ float sm[512];
    const float v = (tid < TP_) ? e[bb * TP_ + tid] : -1e30f;
    sm[tid] = v; __syncthreads();
    for (int o = 256; o > 0; o >>= 1) { if (tid < o) sm[tid] = fmaxf(sm[tid], sm[tid + o]); __syncthreads(); }
    const float mx = sm[0]; __syncthreads();
    const float ev = (tid < TP_) ? expf(v - mx) : 0.f;
    sm[tid] = ev; __syncthreads();
    for (int o = 256; o > 0; o >>= 1) { if (tid < o) sm[tid] += sm[tid + o]; __syncthreads(); }
    const float inv = 1.f / sm[0];
    if (tid < TP_) {
        const float a = ev * inv;
        e[bb * TP_ + tid] = a;
        cov[bb * TP_ + tid] += a;
    }
}

// ctx_vec = attn @ ctx ; writes concat_h = [ctx_vec | h] in f16. grid 32, block 512
__global__ void ctxvec_kernel(const float* __restrict__ attn, const float* __restrict__ context,
                              const float* __restrict__ h, _Float16* __restrict__ concat_h)
{
    const int bb = blockIdx.x, hh = threadIdx.x, b = bb & 7;
    const float* ar = attn + (size_t)bb * TP_;
    float s = 0.f;
    for (int t = 0; t < TP_; ++t)
        s += ar[t] * context[((size_t)(t * B_ + b)) * H_ + hh];
    concat_h[(size_t)bb * 2 * H_ + hh] = (_Float16)s;
    concat_h[(size_t)bb * 2 * H_ + H_ + hh] = (_Float16)h[bb * H_ + hh];
}

// p_gen = sigmoid(cvec.w_p1 + emb.w_p2). grid 32, block 256
__global__ void pgen_kernel(const float* __restrict__ cvec, const float* __restrict__ emb,
                            const float* __restrict__ w_p1, const float* __restrict__ w_p2,
                            float* __restrict__ pgen)
{
    const int bb = blockIdx.x, tid = threadIdx.x;
    float s = cvec[bb * H_ + tid] * w_p1[tid]
            + cvec[bb * H_ + 256 + tid] * w_p1[256 + tid]
            + emb[bb * DW_ + tid] * w_p2[tid];
    __shared__ float sm[256];
    sm[tid] = s; __syncthreads();
    for (int o = 128; o > 0; o >>= 1) { if (tid < o) sm[tid] += sm[tid + o]; __syncthreads(); }
    if (tid == 0) pgen[bb] = 1.f / (1.f + expf(-sm[0]));
}

// softmax over VG (logits already include b_g), in place. grid 32, block 256
__global__ void gen_softmax_kernel(float* __restrict__ gen)
{
    const int bb = blockIdx.x, tid = threadIdx.x;
    float* row = gen + (size_t)bb * VG_;
    __shared__ float sm[256];
    float mx = -1e30f;
    for (int v = tid; v < VG_; v += 256) mx = fmaxf(mx, row[v]);
    sm[tid] = mx; __syncthreads();
    for (int o = 128; o > 0; o >>= 1) { if (tid < o) sm[tid] = fmaxf(sm[tid], sm[tid + o]); __syncthreads(); }
    mx = sm[0]; __syncthreads();
    float s = 0.f;
    for (int v = tid; v < VG_; v += 256) { const float e_ = expf(row[v] - mx); row[v] = e_; s += e_; }
    sm[tid] = s; __syncthreads();
    for (int o = 128; o > 0; o >>= 1) { if (tid < o) sm[tid] += sm[tid + o]; __syncthreads(); }
    const float inv = 1.f / sm[0];
    for (int v = tid; v < VG_; v += 256) row[v] *= inv;
}

// out[bb,v] = p_gen*gen (v<VG) else 0.  grid 6250, block 256
__global__ void mix_base_kernel(const float* __restrict__ gen, const float* __restrict__ pgen,
                                float* __restrict__ out)
{
    const size_t i = (size_t)blockIdx.x * 256 + threadIdx.x;   // < 1.6M
    const int bb = (int)(i / V_TOT), v = (int)(i % V_TOT);
    out[i] = (v < VG_) ? pgen[bb] * gen[(size_t)bb * VG_ + v] : 0.f;
}

// copy distribution: scatter attn via p_ids. grid 50, block 256 (32*400 items)
__global__ void scatter_copy_kernel(const float* __restrict__ attn, const float* __restrict__ pgen,
                                    const int* __restrict__ p_batch, float* __restrict__ out)
{
    const int i = blockIdx.x * 256 + threadIdx.x;
    const int bb = i / TP_, t = i % TP_;
    const int v = p_batch[t * B_ + (bb & 7)];
    atomicAdd(&out[(size_t)bb * V_TOT + v], (1.f - pgen[bb]) * attn[bb * TP_ + t]);
}

// clip 1e-12 then row sum. grid 32, block 256
__global__ void clip_sum_kernel(float* __restrict__ out, float* __restrict__ rowsum)
{
    const int bb = blockIdx.x, tid = threadIdx.x;
    float* row = out + (size_t)bb * V_TOT;
    float s = 0.f;
    for (int v = tid; v < V_TOT; v += 256) {
        const float x = fmaxf(row[v], 1e-12f);
        row[v] = x; s += x;
    }
    __shared__ float sm[256];
    sm[tid] = s; __syncthreads();
    for (int o = 128; o > 0; o >>= 1) { if (tid < o) sm[tid] += sm[tid + o]; __syncthreads(); }
    if (tid == 0) rowsum[bb] = sm[0];
}

// logp = log(out/sum); greedy argmax (first index on ties) -> tok. grid 32, block 256
__global__ void lognorm_argmax_kernel(float* __restrict__ out, const float* __restrict__ rowsum,
                                      int* __restrict__ tok)
{
    const int bb = blockIdx.x, tid = threadIdx.x;
    float* row = out + (size_t)bb * V_TOT;
    const float inv = 1.f / rowsum[bb];
    float best = -1e30f; int bidx = V_TOT;
    for (int v = tid; v < V_TOT; v += 256) {
        const float lp = logf(row[v] * inv);
        row[v] = lp;
        if (lp > best) { best = lp; bidx = v; }
    }
    __shared__ float sv[256]; __shared__ int si[256];
    sv[tid] = best; si[tid] = bidx; __syncthreads();
    for (int o = 128; o > 0; o >>= 1) {
        if (tid < o) {
            if (sv[tid + o] > sv[tid] || (sv[tid + o] == sv[tid] && si[tid + o] < si[tid])) {
                sv[tid] = sv[tid + o]; si[tid] = si[tid + o];
            }
        }
        __syncthreads();
    }
    if (tid == 0) tok[bb] = si[0];
}

// ============================================================================
// Host side
// ============================================================================
extern "C" void kernel_launch(void* const* d_in, const int* in_sizes, int n_in,
                              void* d_out, int out_size, void* d_ws, size_t ws_size,
                              hipStream_t stream)
{
    (void)in_sizes; (void)n_in; (void)out_size; (void)ws_size;
    const int*   p_batch   = (const int*)  d_in[0];
    const float* enc_hidden= (const float*)d_in[1];
    const float* context   = (const float*)d_in[2];
    const float* E         = (const float*)d_in[3];
    const float* W_ih      = (const float*)d_in[4];
    const float* W_hh      = (const float*)d_in[5];
    const float* b_ih      = (const float*)d_in[6];
    const float* b_hh      = (const float*)d_in[7];
    const float* W_q       = (const float*)d_in[8];
    const float* W_k       = (const float*)d_in[9];
    const float* w_cov     = (const float*)d_in[10];
    const float* v_att     = (const float*)d_in[11];
    const float* W_out     = (const float*)d_in[12];
    const float* b_out     = (const float*)d_in[13];
    const float* W_g       = (const float*)d_in[14];
    const float* b_g       = (const float*)d_in[15];
    const float* w_p1      = (const float*)d_in[16];
    const float* w_p2      = (const float*)d_in[17];
    float* out = (float*)d_out;

    // ---- workspace carve-out (all offsets 256B aligned) ----
    char* ws = (char*)d_ws;
    size_t off = 0;
    auto carve = [&](size_t bytes) -> char* {
        char* p = ws + off;
        off = (off + bytes + 255) & ~(size_t)255;
        return p;
    };
    _Float16* Wk_p    = (_Float16*)carve((size_t)H_ * H_ * 2);
    _Float16* WihT_p  = (_Float16*)carve((size_t)DW_ * 4 * H_ * 2);
    _Float16* WhhT_p  = (_Float16*)carve((size_t)H_ * 4 * H_ * 2);
    _Float16* Wq_p    = (_Float16*)carve((size_t)H_ * H_ * 2);
    _Float16* Wout_p  = (_Float16*)carve((size_t)2 * H_ * H_ * 2);
    _Float16* Wg_p    = (_Float16*)carve((size_t)H_ * VG_ * 2);
    _Float16* ctx_h   = (_Float16*)carve((size_t)TP_ * B_ * H_ * 2);
    float*    keys    = (float*)   carve((size_t)TP_ * B_ * H_ * 4);
    float*    h       = (float*)   carve((size_t)BB_ * H_ * 4);
    _Float16* h_h     = (_Float16*)carve((size_t)BB_ * H_ * 2);
    float*    c       = (float*)   carve((size_t)BB_ * H_ * 4);
    float*    cov     = (float*)   carve((size_t)BB_ * TP_ * 4);
    int*      tok     = (int*)     carve((size_t)BB_ * 4);
    float*    emb     = (float*)   carve((size_t)BB_ * DW_ * 4);
    _Float16* emb_h   = (_Float16*)carve((size_t)BB_ * DW_ * 2);
    float*    gates   = (float*)   carve((size_t)BB_ * 4 * H_ * 4);
    float*    qbuf    = (float*)   carve((size_t)BB_ * H_ * 4);
    float*    attn    = (float*)   carve((size_t)BB_ * TP_ * 4);   // energies then attn
    _Float16* concat_h= (_Float16*)carve((size_t)BB_ * 2 * H_ * 2);
    float*    cvec    = (float*)   carve((size_t)BB_ * H_ * 4);
    _Float16* cvec_h  = (_Float16*)carve((size_t)BB_ * H_ * 2);
    float*    gen     = (float*)   carve((size_t)BB_ * VG_ * 4);
    float*    pgen    = (float*)   carve((size_t)BB_ * 4);
    float*    rowsum  = (float*)   carve((size_t)BB_ * 4);

    // ---- one-time precompute: pack weights, cast context, attention keys ----
    pack_b_kernel<<<dim3(H_ / 16, H_ / 32), 32, 0, stream>>>(W_k,   Wk_p,   H_, H_, 0);
    pack_b_kernel<<<dim3(4 * H_ / 16, DW_ / 32), 32, 0, stream>>>(W_ih, WihT_p, DW_, 4 * H_, 1);
    pack_b_kernel<<<dim3(4 * H_ / 16, H_ / 32),  32, 0, stream>>>(W_hh, WhhT_p, H_,  4 * H_, 1);
    pack_b_kernel<<<dim3(H_ / 16, H_ / 32),      32, 0, stream>>>(W_q,  Wq_p,   H_,  H_, 0);
    pack_b_kernel<<<dim3(H_ / 16, 2 * H_ / 32),  32, 0, stream>>>(W_out, Wout_p, 2 * H_, H_, 0);
    pack_b_kernel<<<dim3(VG_ / 16, H_ / 32),     32, 0, stream>>>(W_g,  Wg_p,   H_,  VG_, 0);

    const int nctx = TP_ * B_ * H_;
    f32_to_f16_kernel<<<(nctx + 255) / 256, 256, 0, stream>>>(context, ctx_h, nctx);

    // keys = ctx @ W_k : [3200,512] x [512,512]   (grid: N/16 x M/32)
    wmma_gemm_kernel<<<dim3(H_ / 16, (TP_ * B_) / 32), 32, 0, stream>>>(
        ctx_h, H_, Wk_p, nullptr, keys, H_, H_, 0, nullptr);

    init_state_kernel<<<BB_, H_, 0, stream>>>(enc_hidden, h, h_h, c, cov, tok);

    // ---- decode loop ----
    for (int step = 0; step < STEPS_; ++step) {
        float* outslice = out + (size_t)step * BB_ * V_TOT;

        embed_kernel<<<BB_, DW_, 0, stream>>>(tok, E, emb, emb_h);

        // gates = emb @ W_ih.T + h @ W_hh.T   (M=32 -> one 32-row strip)
        wmma_gemm_kernel<<<dim3(4 * H_ / 16, 1), 32, 0, stream>>>(
            emb_h, DW_, WihT_p, nullptr, gates, 4 * H_, DW_, 0, nullptr);
        wmma_gemm_kernel<<<dim3(4 * H_ / 16, 1), 32, 0, stream>>>(
            h_h, H_, WhhT_p, nullptr, gates, 4 * H_, H_, 1, nullptr);

        lstm_cell_kernel<<<BB_, H_, 0, stream>>>(gates, b_ih, b_hh, h, h_h, c);

        // q = h_new @ W_q
        wmma_gemm_kernel<<<dim3(H_ / 16, 1), 32, 0, stream>>>(
            h_h, H_, Wq_p, nullptr, qbuf, H_, H_, 0, nullptr);

        attn_energy_kernel<<<dim3(TP_, BB_), 256, 0, stream>>>(qbuf, keys, cov, w_cov, v_att, attn);
        attn_softmax_kernel<<<BB_, 512, 0, stream>>>(attn, cov);

        ctxvec_kernel<<<BB_, H_, 0, stream>>>(attn, context, h, concat_h);

        // cvec = tanh(concat @ W_out + b_out)  (fused tanh + f16 copy)
        wmma_gemm_kernel<<<dim3(H_ / 16, 1), 32, 0, stream>>>(
            concat_h, 2 * H_, Wout_p, b_out, cvec, H_, 2 * H_, 4 | 2 | 8, cvec_h);

        pgen_kernel<<<BB_, 256, 0, stream>>>(cvec, emb, w_p1, w_p2, pgen);

        // gen logits = cvec @ W_g + b_g   [32,512]x[512,20000]
        wmma_gemm_kernel<<<dim3(VG_ / 16, 1), 32, 0, stream>>>(
            cvec_h, H_, Wg_p, b_g, gen, VG_, H_, 4, nullptr);
        gen_softmax_kernel<<<BB_, 256, 0, stream>>>(gen);

        mix_base_kernel<<<(BB_ * V_TOT) / 256, 256, 0, stream>>>(gen, pgen, outslice);
        scatter_copy_kernel<<<(BB_ * TP_) / 256, 256, 0, stream>>>(attn, pgen, p_batch, outslice);
        clip_sum_kernel<<<BB_, 256, 0, stream>>>(outslice, rowsum);
        lognorm_argmax_kernel<<<BB_, 256, 0, stream>>>(outslice, rowsum, tok);
    }
}
